// GAT_16776142258595
// MI455X (gfx1250) — matrix operand order repaired and checked by reading, not compiled, not run on hardware
//
#include <hip/hip_runtime.h>
#include <hip/hip_bf16.h>
#include <stdint.h>

// ---------------- CDNA5 WMMA types ----------------
typedef __attribute__((ext_vector_type(16))) __bf16 v16bf;
typedef __attribute__((ext_vector_type(8)))  float  v8f;

#define BATCH 2
#define DIN   128
#define DOUT  128   // H*hid
#define HEADS 4
#define HID   32
#define NEG_SLOPE 0.2f
#define MTILES_PER_BLOCK 8

// Native CDNA5 f32 atomic max (no return -> STOREcnt), device scope so it
// combines correctly across WGPs in L2.
__device__ __forceinline__ void atomicMaxF32(float* p, float v) {
  asm volatile("global_atomic_max_num_f32 %0, %1, off scope:SCOPE_DEV"
               : : "v"(p), "v"(v) : "memory");
}

__device__ __forceinline__ float leaky(float v) {
  return v > 0.0f ? v : NEG_SLOPE * v;
}

// ---------------- init: zero accumulators, -inf maxima, bias-init output ----
__global__ void k_init(float* __restrict__ rst1,
                       float* __restrict__ emax1, float* __restrict__ denom1,
                       float* __restrict__ emax2, float* __restrict__ denom2,
                       float* __restrict__ out, const float* __restrict__ b2,
                       int NB) {
  int t = blockIdx.x * blockDim.x + threadIdx.x;
  if (t < NB * DOUT) rst1[t] = 0.0f;
  if (t < NB * HEADS) { emax1[t] = -3.402823466e38f; denom1[t] = 0.0f; }
  if (t < NB)         { emax2[t] = -3.402823466e38f; denom2[t] = 0.0f;
                        out[t]   = b2[0]; }
}

// ---------------- layer1 GEMM: z1[NB,128] = X[NB,128] @ W1[128,128] --------
// X row r = h[b][n][:] with n = r/2, b = r&1 (the swapaxes fold).
// Each wave owns one N-tile (wave id in block) and keeps the 4 K-step B
// fragments of W1 resident in VGPRs, sweeping 8 M-tiles: 32 v_wmma per wave,
// W1 loaded once per wave instead of once per tile.
__global__ void k_gemm1(const float* __restrict__ h,
                        const float* __restrict__ W1,
                        float* __restrict__ z1,
                        int N, int NB) {
  int lane  = threadIdx.x & 31;
  int ntile = threadIdx.x >> 5;                 // 8 waves = 8 N-tiles
  int mtiles = (NB + 15) >> 4;
  int mbase  = blockIdx.x * MTILES_PER_BLOCK;

  int half = lane >> 4;                         // 0: lanes 0-15, 1: 16-31
  int l15  = lane & 15;
  int bcol = ntile * 16 + l15;

  // B (32x16 bf16 per K-step): elems 0..7 -> K = kk*32 + half*8 + i, elems 8..15 -> K = kk*32 + half*8 + 16 + (i-8)
  v16bf bfrag[4];
#pragma unroll
  for (int kk = 0; kk < 4; ++kk) {
    int bk = kk * 32 + half * 8;
#pragma unroll
    for (int i = 0; i < 16; ++i)
      bfrag[kk][i] = (__bf16)W1[(size_t)(bk + i + (i >= 8 ? 8 : 0)) * DOUT + bcol];
  }

#pragma unroll
  for (int m = 0; m < MTILES_PER_BLOCK; ++m) {
    int mtile = mbase + m;
    if (mtile < mtiles) {
      int arow = mtile * 16 + l15;
      if (arow >= NB) arow = NB - 1;            // clamp (NB % 16 == 0 here)
      int nn = arow >> 1, bb = arow & 1;
      const float* aptr = h + ((size_t)bb * N + nn) * DIN;

      // prefetch next M-tile's A row into cache (global_prefetch)
      if (mtile + 1 < mtiles) {
        int arow2 = (mtile + 1) * 16 + l15;
        if (arow2 >= NB) arow2 = NB - 1;
        const float* p2 = h + ((size_t)(arow2 & 1) * N + (arow2 >> 1)) * DIN;
        __builtin_prefetch(p2, 0, 1);
      }

      v8f c = {};
#pragma unroll
      for (int kk = 0; kk < 4; ++kk) {
        // A (16x32 bf16): elems 0..7  -> K = kk*32 + half*8 + i
        //                 elems 8..15 -> K = kk*32 + half*8 + 16 + i
        int ak = kk * 32 + half * 8;
        v16bf a;
#pragma unroll
        for (int i = 0; i < 8; ++i) {
          a[i]     = (__bf16)aptr[ak + i];
          a[i + 8] = (__bf16)aptr[ak + 16 + i];
        }
        c = __builtin_amdgcn_wmma_f32_16x16x32_bf16(false, a, false, bfrag[kk],
                                                    (short)0, c, false, false);
      }
      // C layout: elem r -> row = mtile*16 + half*8 + r, col = ntile*16 + l15
      int rbase = mtile * 16 + half * 8;
#pragma unroll
      for (int r = 0; r < 8; ++r) {
        int row = rbase + r;
        if (row < NB) z1[(size_t)row * DOUT + ntile * 16 + l15] = c[r];
      }
    }
  }
}

// ---------------- attention coefficients: el/er[NB,H] -----------------------
__global__ void k_coeff1(const float* __restrict__ z1,
                         const float* __restrict__ al1,
                         const float* __restrict__ ar1,
                         float* __restrict__ el1, float* __restrict__ er1,
                         int NB) {
  int t = blockIdx.x * blockDim.x + threadIdx.x;
  if (t >= NB * HEADS) return;
  int row = t >> 2, hh = t & 3;
  const float* zp  = z1 + (size_t)row * DOUT + hh * HID;
  const float* alp = al1 + hh * HID;
  const float* arp = ar1 + hh * HID;
  float sl = 0.0f, sr = 0.0f;
#pragma unroll 8
  for (int d = 0; d < HID; ++d) {
    float zz = zp[d];
    sl += zz * alp[d];
    sr += zz * arp[d];
  }
  el1[t] = sl;
  er1[t] = sr;
}

// ---------------- layer1 edge softmax: pass 1 (segment max) -----------------
__global__ void k_edge_max1(const int* __restrict__ src, const int* __restrict__ dst,
                            const float* __restrict__ el1, const float* __restrict__ er1,
                            float* __restrict__ emax1, int E) {
  int t = blockIdx.x * blockDim.x + threadIdx.x;
  if (t >= E * BATCH * HEADS) return;
  int e = t >> 3, bh = t & 7;               // bh = b*4 + h
  int s = src[e], d = dst[e];
  float v = leaky(el1[s * 8 + bh] + er1[d * 8 + bh]);
  atomicMaxF32(&emax1[d * 8 + bh], v);
}

// ---------------- layer1 edge softmax: pass 2 (denominator) -----------------
__global__ void k_edge_den1(const int* __restrict__ src, const int* __restrict__ dst,
                            const float* __restrict__ el1, const float* __restrict__ er1,
                            const float* __restrict__ emax1, float* __restrict__ denom1,
                            int E) {
  int t = blockIdx.x * blockDim.x + threadIdx.x;
  if (t >= E * BATCH * HEADS) return;
  int e = t >> 3, bh = t & 7;
  int s = src[e], d = dst[e];
  float v = leaky(el1[s * 8 + bh] + er1[d * 8 + bh]);
  atomicAdd(&denom1[d * 8 + bh], __expf(v - emax1[d * 8 + bh]));
}

// ---------------- layer1 aggregation: wave per (edge,batch) -----------------
// 32 lanes cover the 128 feature columns in 4 strides -> coalesced gather of
// z1[src] row and coalesced global_atomic_add_f32 into rst1[dst] row.
__global__ void k_edge_aggr1(const int* __restrict__ src, const int* __restrict__ dst,
                             const float* __restrict__ el1, const float* __restrict__ er1,
                             const float* __restrict__ emax1, const float* __restrict__ denom1,
                             const float* __restrict__ z1, float* __restrict__ rst1,
                             int E) {
  int gt = blockIdx.x * blockDim.x + threadIdx.x;
  int wave = gt >> 5, lane = gt & 31;
  if (wave >= E * BATCH) return;
  int e = wave >> 1, b = wave & 1;
  int s = src[e], d = dst[e];
  const float* zrow = z1 + (size_t)(s * 2 + b) * DOUT;
  float* orow = rst1 + (size_t)(d * 2 + b) * DOUT;
#pragma unroll
  for (int k = 0; k < HEADS; ++k) {
    int bh = b * 4 + k;
    float v = leaky(el1[s * 8 + bh] + er1[d * 8 + bh]);
    float alpha = __expf(v - emax1[d * 8 + bh]) / denom1[d * 8 + bh];
    int c = k * 32 + lane;
    atomicAdd(&orow[c], alpha * zrow[c]);
  }
}

// ---------------- layer2 node: z2 = dot(elu(rst1+b1), W2), wave per row -----
__global__ void k_node2(const float* __restrict__ rst1, const float* __restrict__ b1,
                        const float* __restrict__ W2, float* __restrict__ z2,
                        int NB) {
  int gt = blockIdx.x * blockDim.x + threadIdx.x;
  int wave = gt >> 5, lane = gt & 31;
  if (wave >= NB) return;
  const float* r = rst1 + (size_t)wave * DOUT;
  float acc = 0.0f;
#pragma unroll
  for (int k = 0; k < 4; ++k) {
    int c = k * 32 + lane;
    float x = r[c] + b1[c];
    x = x > 0.0f ? x : (__expf(x) - 1.0f);   // ELU
    acc += x * W2[c];
  }
#pragma unroll
  for (int off = 16; off; off >>= 1) acc += __shfl_xor(acc, off, 32);
  if (lane == 0) z2[wave] = acc;
}

// ---------------- layer2 edge softmax (scalar features) ---------------------
__global__ void k_edge_max2(const int* __restrict__ src, const int* __restrict__ dst,
                            const float* __restrict__ z2,
                            const float* __restrict__ al2, const float* __restrict__ ar2,
                            float* __restrict__ emax2, int E) {
  int t = blockIdx.x * blockDim.x + threadIdx.x;
  if (t >= E * BATCH) return;
  int e = t >> 1, b = t & 1;
  int s = src[e], d = dst[e];
  float v = leaky(z2[s * 2 + b] * al2[0] + z2[d * 2 + b] * ar2[0]);
  atomicMaxF32(&emax2[d * 2 + b], v);
}

__global__ void k_edge_den2(const int* __restrict__ src, const int* __restrict__ dst,
                            const float* __restrict__ z2,
                            const float* __restrict__ al2, const float* __restrict__ ar2,
                            const float* __restrict__ emax2, float* __restrict__ denom2,
                            int E) {
  int t = blockIdx.x * blockDim.x + threadIdx.x;
  if (t >= E * BATCH) return;
  int e = t >> 1, b = t & 1;
  int s = src[e], d = dst[e];
  float v = leaky(z2[s * 2 + b] * al2[0] + z2[d * 2 + b] * ar2[0]);
  atomicAdd(&denom2[d * 2 + b], __expf(v - emax2[d * 2 + b]));
}

// out[b*N + d] was pre-initialized to b2[0]; accumulate alpha * z2[src].
__global__ void k_edge_aggr2(const int* __restrict__ src, const int* __restrict__ dst,
                             const float* __restrict__ z2,
                             const float* __restrict__ al2, const float* __restrict__ ar2,
                             const float* __restrict__ emax2, const float* __restrict__ denom2,
                             float* __restrict__ out, int N, int E) {
  int t = blockIdx.x * blockDim.x + threadIdx.x;
  if (t >= E * BATCH) return;
  int e = t >> 1, b = t & 1;
  int s = src[e], d = dst[e];
  float zs = z2[s * 2 + b];
  float v = leaky(zs * al2[0] + z2[d * 2 + b] * ar2[0]);
  float alpha = __expf(v - emax2[d * 2 + b]) / denom2[d * 2 + b];
  atomicAdd(&out[(size_t)b * N + d], alpha * zs);
}

// ---------------- launch ----------------------------------------------------
extern "C" void kernel_launch(void* const* d_in, const int* in_sizes, int n_in,
                              void* d_out, int out_size, void* d_ws, size_t ws_size,
                              hipStream_t stream) {
  const float* h   = (const float*)d_in[0];
  const int*   src = (const int*)d_in[1];
  const int*   dst = (const int*)d_in[2];
  const float* W1  = (const float*)d_in[3];
  const float* al1 = (const float*)d_in[4];
  const float* ar1 = (const float*)d_in[5];
  const float* b1  = (const float*)d_in[6];
  const float* W2  = (const float*)d_in[7];
  const float* al2 = (const float*)d_in[8];
  const float* ar2 = (const float*)d_in[9];
  const float* b2  = (const float*)d_in[10];

  const int E  = in_sizes[1];
  const int N  = in_sizes[0] / (BATCH * DIN);
  const int NB = BATCH * N;

  float* ws     = (float*)d_ws;
  float* z1     = ws;                               // NB*128
  float* rst1   = z1     + (size_t)NB * DOUT;       // NB*128
  float* el1    = rst1   + (size_t)NB * DOUT;       // NB*4
  float* er1    = el1    + (size_t)NB * HEADS;      // NB*4
  float* emax1  = er1    + (size_t)NB * HEADS;      // NB*4
  float* denom1 = emax1  + (size_t)NB * HEADS;      // NB*4
  float* z2     = denom1 + (size_t)NB * HEADS;      // NB
  float* emax2  = z2     + (size_t)NB;              // NB
  float* denom2 = emax2  + (size_t)NB;              // NB
  float* out    = (float*)d_out;                    // B*N

  const int TPB = 256;
  int g;

  g = (NB * DOUT + TPB - 1) / TPB;
  k_init<<<g, TPB, 0, stream>>>(rst1, emax1, denom1, emax2, denom2, out, b2, NB);

  int mtiles = (NB + 15) >> 4;
  int gblocks = (mtiles + MTILES_PER_BLOCK - 1) / MTILES_PER_BLOCK;
  k_gemm1<<<gblocks, TPB, 0, stream>>>(h, W1, z1, N, NB);

  g = (NB * HEADS + TPB - 1) / TPB;
  k_coeff1<<<g, TPB, 0, stream>>>(z1, al1, ar1, el1, er1, NB);

  g = (E * BATCH * HEADS + TPB - 1) / TPB;
  k_edge_max1<<<g, TPB, 0, stream>>>(src, dst, el1, er1, emax1, E);
  k_edge_den1<<<g, TPB, 0, stream>>>(src, dst, el1, er1, emax1, denom1, E);

  g = (int)(((size_t)E * BATCH * 32 + TPB - 1) / TPB);
  k_edge_aggr1<<<g, TPB, 0, stream>>>(src, dst, el1, er1, emax1, denom1, z1, rst1, E);

  g = (NB * 32 + TPB - 1) / TPB;
  k_node2<<<g, TPB, 0, stream>>>(rst1, b1, W2, z2, NB);

  g = (E * BATCH + TPB - 1) / TPB;
  k_edge_max2<<<g, TPB, 0, stream>>>(src, dst, z2, al2, ar2, emax2, E);
  k_edge_den2<<<g, TPB, 0, stream>>>(src, dst, z2, al2, ar2, emax2, denom2, E);
  k_edge_aggr2<<<g, TPB, 0, stream>>>(src, dst, z2, al2, ar2, emax2, denom2, out, N, E);
}
